// AdditiveAttention_49323404428002
// MI455X (gfx1250) — matrix operand order repaired and checked by reading, not compiled, not run on hardware
//
#include <hip/hip_runtime.h>
#include <hip/hip_bf16.h>
#include <math.h>

#define B_DIM 16
#define S_DIM 2048
#define D_DIM 512
#define A_DIM 512
#define ROWPAD 520   // bf16 elems per LDS row (16B-aligned stride, avoids bank clash)
#define RT 4         // row-tiles (of 16 rows) per block => 64 rows/block

typedef __attribute__((ext_vector_type(16))) __bf16 v16bf;
typedef __attribute__((ext_vector_type(8)))  __bf16 v8bf;
typedef __attribute__((ext_vector_type(8)))  float  v8f;

// ---------------------------------------------------------------------------
// Kernel 0: repack Wq,Wk (f32, row-major [D,A]) -> bf16 in WMMA-B-fragment
// order. Fragment (mat, nt, ks): lane L holds 16 contiguous bf16 with
//   col = nt*16 + L%16,  K = ks*32 + (L/16)*16 + e   (e = 0..15)
// Linear index g = ((mat*32 + nt)*16 + ks)*32 + L ; lane data at packed[g*16].
// ---------------------------------------------------------------------------
__global__ __launch_bounds__(256) void pack_kernel(
    const float* __restrict__ Wq, const float* __restrict__ Wk,
    __bf16* __restrict__ packed)
{
    int g    = blockIdx.x * 256 + threadIdx.x;   // 0 .. 32767
    int lane = g & 31;
    int ks   = (g >> 5)  & 15;
    int nt   = (g >> 9)  & 31;
    int mat  = (g >> 14) & 1;
    const float* W = mat ? Wk : Wq;
    int colA = nt * 16 + (lane & 15);
    int kb   = ks * 32 + (lane >> 4) * 16;
    __bf16* out = packed + (size_t)g * 16;
#pragma unroll
    for (int e = 0; e < 16; ++e)
        out[e] = (__bf16)W[(size_t)(kb + e) * A_DIM + colA];
}

// ---------------------------------------------------------------------------
// Kernel 1: scores[b,s] = sum_a tanh( (query@Wq)[s,a] + (key@Wk)[s,a] ) * Ws[a]
// One block = 64 rows (4 row-tiles). 8 waves x 4 N-tiles; each B fragment is
// reused across 4 row-tiles (4 accumulators) -> 4x less L2 weight traffic.
// K loop runs query then key into the SAME accumulators (q+k for free).
// bf16 WMMA, f32 accumulate. Dynamic LDS: sQ(64xROWPAD) | sK(64xROWPAD) | 64f.
// ---------------------------------------------------------------------------
__global__ __launch_bounds__(256) void scores_kernel(
    const float* __restrict__ query, const float* __restrict__ key_,
    const __bf16* __restrict__ packedW, const float* __restrict__ Ws,
    float* __restrict__ scores)
{
    extern __shared__ char smem[];
    __bf16* sQ     = (__bf16*)smem;
    __bf16* sK     = sQ + 64 * ROWPAD;
    float*  sScore = (float*)(sK + 64 * ROWPAD);

    const int tid   = threadIdx.x;
    const int blk   = blockIdx.x;                       // over B*S/64
    const size_t gb = (size_t)blk * 64 * D_DIM;

    // Stage 64 rows of query & key into LDS as bf16 (float4 loads, coalesced).
    for (int i = tid; i < 64 * D_DIM / 4; i += 256) {
        int r = (i * 4) / D_DIM;
        int c = (i * 4) % D_DIM;
        float4 q4 = ((const float4*)(query + gb))[i];
        float4 k4 = ((const float4*)(key_  + gb))[i];
        __bf16* dq = &sQ[r * ROWPAD + c];
        __bf16* dk = &sK[r * ROWPAD + c];
        dq[0] = (__bf16)q4.x; dq[1] = (__bf16)q4.y;
        dq[2] = (__bf16)q4.z; dq[3] = (__bf16)q4.w;
        dk[0] = (__bf16)k4.x; dk[1] = (__bf16)k4.y;
        dk[2] = (__bf16)k4.z; dk[3] = (__bf16)k4.w;
    }
    if (tid < 64) sScore[tid] = 0.0f;
    __syncthreads();

    const int wave = tid >> 5;
    const int lane = tid & 31;
    const int col  = lane & 15;       // A-frag row-in-tile == B-frag col == C col
    const int half = lane >> 4;
    const int kbA  = half * 16;       // A fragment K base within a 32-slice

    float sacc[RT][8];
#pragma unroll
    for (int rt = 0; rt < RT; ++rt)
#pragma unroll
        for (int i = 0; i < 8; ++i) sacc[rt][i] = 0.0f;

#pragma unroll 1
    for (int t = 0; t < 4; ++t) {
        const int nt = wave * 4 + t;
        v8f c[RT];
#pragma unroll
        for (int rt = 0; rt < RT; ++rt) c[rt] = (v8f){};

#pragma unroll 1
        for (int mat = 0; mat < 2; ++mat) {
            const __bf16* src   = mat ? sK : sQ;
            const __bf16* wbase = packedW +
                (((size_t)(mat * 32 + nt) * 16) * 32 + lane) * 16;
#pragma unroll 4
            for (int ks = 0; ks < 16; ++ks) {
                // B fragment: 32 contiguous bytes per lane, reused for RT tiles
                const __bf16* bp = wbase + (size_t)ks * (32 * 16);
                v8bf b_lo = *(const v8bf*)(bp);        // global_load_b128
                v8bf b_hi = *(const v8bf*)(bp + 8);    // global_load_b128
                v16bf bfrag = __builtin_shufflevector(b_lo, b_hi,
                    0,1,2,3,4,5,6,7,8,9,10,11,12,13,14,15);
#pragma unroll
                for (int rt = 0; rt < RT; ++rt) {
                    // A fragment: row = rt*16 + lane%16; elems 0-15 -> K=kbA..+15
                    // (contiguous, matches packed B)
                    const __bf16* ap =
                        &src[(rt * 16 + col) * ROWPAD + ks * 32 + kbA];
                    v8bf a_lo = *(const v8bf*)(ap);        // ds_load_b128
                    v8bf a_hi = *(const v8bf*)(ap + 8);    // ds_load_b128
                    v16bf afrag = __builtin_shufflevector(a_lo, a_hi,
                        0,1,2,3,4,5,6,7,8,9,10,11,12,13,14,15);
                    c[rt] = __builtin_amdgcn_wmma_f32_16x16x32_bf16(
                                false, afrag, false, bfrag,
                                (short)0, c[rt], false, false);
                }
            }
        }
        // C layout: lane holds rows i + 8*half of the tile, column = col
        const float wsv = Ws[nt * 16 + col];
#pragma unroll
        for (int rt = 0; rt < RT; ++rt)
#pragma unroll
            for (int i = 0; i < 8; ++i)
                sacc[rt][i] += tanhf(c[rt][i]) * wsv;
    }

    // Per-row reduction across columns/waves via LDS float atomics (ds_add_f32)
#pragma unroll
    for (int rt = 0; rt < RT; ++rt)
#pragma unroll
        for (int i = 0; i < 8; ++i)
            atomicAdd(&sScore[rt * 16 + i + 8 * half], sacc[rt][i]);
    __syncthreads();

    if (tid < 64) scores[(size_t)blk * 64 + tid] = sScore[tid];
}

// ---------------------------------------------------------------------------
// Kernel 2: softmax over S per batch. weights -> d_out[B*A ..]
// ---------------------------------------------------------------------------
__global__ __launch_bounds__(256) void softmax_kernel(
    const float* __restrict__ scores, float* __restrict__ weights)
{
    __shared__ float red[256];
    const int b = blockIdx.x, tid = threadIdx.x;
    const float* s = scores + (size_t)b * S_DIM;

    float m = -INFINITY;
    for (int i = tid; i < S_DIM; i += 256) m = fmaxf(m, s[i]);
    red[tid] = m; __syncthreads();
    for (int off = 128; off > 0; off >>= 1) {
        if (tid < off) red[tid] = fmaxf(red[tid], red[tid + off]);
        __syncthreads();
    }
    m = red[0]; __syncthreads();

    float sum = 0.0f;
    for (int i = tid; i < S_DIM; i += 256) sum += __expf(s[i] - m);
    red[tid] = sum; __syncthreads();
    for (int off = 128; off > 0; off >>= 1) {
        if (tid < off) red[tid] += red[tid + off];
        __syncthreads();
    }
    const float inv = 1.0f / red[0];
    for (int i = tid; i < S_DIM; i += 256)
        weights[(size_t)b * S_DIM + i] = __expf(s[i] - m) * inv;
}

// ---------------------------------------------------------------------------
// Kernel 3: t[b,d] = sum_s w[b,s] * value[b,s,d]   (memory-bound stream)
// grid = B*8 blocks; block covers 64 cols x 4-way s-split.
// ---------------------------------------------------------------------------
__global__ __launch_bounds__(256) void vreduce_kernel(
    const float* __restrict__ value, const float* __restrict__ weights,
    float* __restrict__ t)
{
    __shared__ float part[256];
    const int b     = blockIdx.x >> 3;
    const int chunk = blockIdx.x & 7;
    const int tid   = threadIdx.x;
    const int col   = chunk * 64 + (tid & 63);
    const int phase = tid >> 6;

    const float* w = weights + (size_t)b * S_DIM;
    float acc = 0.0f;
    for (int s = phase; s < S_DIM; s += 4)
        acc += w[s] * value[((size_t)b * S_DIM + s) * D_DIM + col];
    part[tid] = acc; __syncthreads();
    if (tid < 64)
        t[(size_t)b * D_DIM + chunk * 64 + tid] =
            part[tid] + part[tid + 64] + part[tid + 128] + part[tid + 192];
}

// ---------------------------------------------------------------------------
// Kernel 4: context[b,a] = sum_d t[b,d] * Wv[d,a]   (tiny, ~8 MFLOP)
// ---------------------------------------------------------------------------
__global__ __launch_bounds__(256) void context_kernel(
    const float* __restrict__ t, const float* __restrict__ Wv,
    float* __restrict__ context)
{
    const int g = blockIdx.x * 256 + threadIdx.x;  // 0 .. B*A-1
    const int b = g / A_DIM;
    const int a = g % A_DIM;
    float acc = 0.0f;
#pragma unroll 4
    for (int d = 0; d < D_DIM; ++d)
        acc = fmaf(t[(size_t)b * D_DIM + d], Wv[(size_t)d * A_DIM + a], acc);
    context[g] = acc;
}

// ---------------------------------------------------------------------------
extern "C" void kernel_launch(void* const* d_in, const int* in_sizes, int n_in,
                              void* d_out, int out_size, void* d_ws, size_t ws_size,
                              hipStream_t stream) {
    const float* query = (const float*)d_in[0];
    const float* key_  = (const float*)d_in[1];
    const float* value = (const float*)d_in[2];
    const float* Wq    = (const float*)d_in[3];
    const float* Wk    = (const float*)d_in[4];
    const float* Wv    = (const float*)d_in[5];
    const float* Ws    = (const float*)d_in[6];

    float* out_context = (float*)d_out;                    // [B, A]
    float* out_weights = (float*)d_out + B_DIM * A_DIM;    // [B, S, 1]

    // Workspace layout
    char* ws = (char*)d_ws;
    __bf16* packedW = (__bf16*)ws;                               // 1 MB
    float*  scores  = (float*)(ws + (1u << 20));                 // 128 KB
    float*  tvec    = (float*)(ws + (1u << 20) + B_DIM * S_DIM * 4); // 32 KB

    // Dynamic LDS for scores_kernel: 2 x (64 x ROWPAD) bf16 + 64 floats
    const size_t smem_bytes = (size_t)2 * 64 * ROWPAD * sizeof(__bf16)
                            + 64 * sizeof(float);

    pack_kernel   <<<128, 256, 0, stream>>>(Wq, Wk, packedW);
    scores_kernel <<<(B_DIM * S_DIM) / 64, 256, smem_bytes, stream>>>(
                      query, key_, packedW, Ws, scores);
    softmax_kernel<<<B_DIM, 256, 0, stream>>>(scores, out_weights);
    vreduce_kernel<<<B_DIM * 8, 256, 0, stream>>>(value, out_weights, tvec);
    context_kernel<<<(B_DIM * A_DIM) / 256, 256, 0, stream>>>(
                      tvec, Wv, out_context);
}
